// ChebNet_64991445123422
// MI455X (gfx1250) — matrix operand order repaired and checked by reading, not compiled, not run on hardware
//
#include <hip/hip_runtime.h>
#include <math.h>

typedef __attribute__((ext_vector_type(16))) _Float16 v16h;
typedef __attribute__((ext_vector_type(8)))  float    v8f;

#define NPG    100      // nodes per graph
#define PADR   112      // padded rows (7 * 16)
#define H      64
#define BLOCK  256
#define NWAVES 8
#define NTILES 28       // 7 M-tiles * 4 N-tiles

// packed-B sizes: per layer 5k * 2kk * 4nt * 32lane * 16e halves
#define PKB_HALVES 20480
#define PKB_BYTES  (PKB_HALVES * 2)
#define EDGE_OFS   (2 * PKB_BYTES)   // 81920 bytes, 8-aligned

// ---------------------------------------------------------------------------
// Init kernel: pack W2/W3 (5,64,64 f32) into f16 in exact WMMA B-fragment
// per-lane order, so the main loop loads one v16h (32B) per fragment.
// ISA 05_wmma B layout: lane -> n, half = lane>>4; element e -> K row:
//   e<8 : K = e + 8*half ;  e>=8 : K = e + 8 + 8*half   (per 32-wide K block)
// ---------------------------------------------------------------------------
__global__ void pack_weights(const float* __restrict__ W2,
                             const float* __restrict__ W3,
                             _Float16* __restrict__ pk) {
    int idx = blockIdx.x * blockDim.x + threadIdx.x;
    if (idx >= 2 * PKB_HALVES) return;
    int layer = idx / PKB_HALVES;
    int r = idx % PKB_HALVES;
    int k  = r / 4096; r %= 4096;
    int kk = r / 2048; r %= 2048;
    int nt = r / 512;  r %= 512;
    int lane = r / 16;
    int e    = r % 16;
    int n    = nt * 16 + (lane & 15);
    int half = lane >> 4;
    int kl   = kk * 32 + ((e < 8) ? (e + 8 * half) : (e + 8 + 8 * half));
    const float* W = layer ? W3 : W2;
    pk[idx] = (_Float16)W[k * 4096 + kl * 64 + n];
}

// Load one 16x32 f16 A-fragment slice from an fp32 LDS row (two 8-float runs).
__device__ __forceinline__ v16h load_afrag(const float* __restrict__ row, int half) {
    const float4* p0 = (const float4*)(row + 8 * half);
    const float4* p1 = (const float4*)(row + 16 + 8 * half);
    float4 r0 = p0[0], r1 = p0[1], r2 = p1[0], r3 = p1[1];
    v16h a;
    a[0]  = (_Float16)r0.x; a[1]  = (_Float16)r0.y;
    a[2]  = (_Float16)r0.z; a[3]  = (_Float16)r0.w;
    a[4]  = (_Float16)r1.x; a[5]  = (_Float16)r1.y;
    a[6]  = (_Float16)r1.z; a[7]  = (_Float16)r1.w;
    a[8]  = (_Float16)r2.x; a[9]  = (_Float16)r2.y;
    a[10] = (_Float16)r2.z; a[11] = (_Float16)r2.w;
    a[12] = (_Float16)r3.x; a[13] = (_Float16)r3.y;
    a[14] = (_Float16)r3.z; a[15] = (_Float16)r3.w;
    return a;
}

// accE/accO += (f16)cur(112x64) @ (f16)Wk.  Tiles t = swave + 8i; nt = swave&3
// is constant per wave so only 2 B fragments are needed per k.  Tiles 0..2 are
// unconditionally valid; tile 3 is guarded by a *scalar* branch.  Even/odd
// accumulators keep the two wmmas per tile on independent dependency chains.
__device__ __forceinline__ void wmma_term(const float* __restrict__ cur,
                                          const v16h* __restrict__ packBk,
                                          v8f accE[4], v8f accO[4],
                                          int swave, int lane) {
    const int m0 = lane & 15, half = lane >> 4;
    const int ntw = swave & 3;
    const int mtb = swave >> 2;
    v16h bf0 = packBk[(0 * 4 + ntw) * 32 + lane];
    v16h bf1 = packBk[(1 * 4 + ntw) * 32 + lane];
#pragma unroll
    for (int i = 0; i < 3; ++i) {
        const float* row = cur + ((mtb + 2 * i) * 16 + m0) * H;
        v16h a0 = load_afrag(row, half);
        v16h a1 = load_afrag(row + 32, half);
        accE[i] = __builtin_amdgcn_wmma_f32_16x16x32_f16(
                      false, a0, false, bf0, (short)0, accE[i], false, false);
        accO[i] = __builtin_amdgcn_wmma_f32_16x16x32_f16(
                      false, a1, false, bf1, (short)0, accO[i], false, false);
    }
    if (swave < 4) {   // tile 3: mt = 6 (only lower waves)
        const float* row = cur + (6 * 16 + m0) * H;
        v16h a0 = load_afrag(row, half);
        v16h a1 = load_afrag(row + 32, half);
        accE[3] = __builtin_amdgcn_wmma_f32_16x16x32_f16(
                      false, a0, false, bf0, (short)0, accE[3], false, false);
        accO[3] = __builtin_amdgcn_wmma_f32_16x16x32_f16(
                      false, a1, false, bf1, (short)0, accO[3], false, false);
    }
}

// dst[i,f] = scale * (dst[i,f] + cl*src[i,f] + sum_edges coef * src[col,f])
// CSR gather: no atomics, one writer per element.
__device__ __forceinline__ void gather64(float* __restrict__ dst,
                                         const float* __restrict__ src,
                                         float cl, float scale,
                                         const int2* __restrict__ eg,
                                         const int* __restrict__ rowstart, int tid) {
    int f = tid & 63;
    for (int i = tid >> 6; i < NPG; i += 4) {
        float v = dst[i * H + f] + cl * src[i * H + f];
        int e0 = rowstart[i], e1 = rowstart[i + 1];
        for (int idx = e0; idx < e1; ++idx) {
            int2 ed = eg[idx];
            v += __int_as_float(ed.y) * src[ed.x * H + f];
        }
        dst[i * H + f] = scale * v;
    }
}

__device__ __forceinline__ void gather3(float* __restrict__ dst,
                                        const float* __restrict__ src,
                                        float cl, float scale,
                                        const int2* __restrict__ eg,
                                        const int* __restrict__ rowstart, int tid) {
    int ch = tid & 3;   // ch 3 operates on zero padding lane, harmless
    for (int i = tid >> 2; i < NPG; i += 64) {
        float v = dst[i * 4 + ch] + cl * src[i * 4 + ch];
        int e0 = rowstart[i], e1 = rowstart[i + 1];
        for (int idx = e0; idx < e1; ++idx) {
            int2 ed = eg[idx];
            v += __int_as_float(ed.y) * src[ed.x * 4 + ch];
        }
        dst[i * 4 + ch] = scale * v;
    }
}

// Cheb layer H=64 -> H=64: input in bufA, relu(conv) output back in bufA.
__device__ void cheb_layer64(float* bufA, float* bufB,
                             const int2* eg, const int* rowstart, float cl,
                             const v16h* packB, const float* __restrict__ bias) {
    const int tid = threadIdx.x;
    const int lane = tid & 31;
    const int swave = __builtin_amdgcn_readfirstlane(tid >> 5);
    v8f accE[4] = {};
    v8f accO[4] = {};

    float* cur = bufA;
    float* oth = bufB;
    for (int k = 0; k < 5; ++k) {
        if (k == 1) {                       // Tx1 = Lhat(Tx0)
            for (int i = tid; i < PADR * H; i += BLOCK) oth[i] = 0.0f;
            __syncthreads();
            gather64(oth, cur, cl, 1.0f, eg, rowstart, tid);
            __syncthreads();
            float* t = cur; cur = oth; oth = t;
        } else if (k >= 2) {                // Tx_k = 2*Lhat(Tx_{k-1}) - Tx_{k-2}
            for (int i = tid; i < PADR * H; i += BLOCK) oth[i] *= -0.5f;
            __syncthreads();
            gather64(oth, cur, cl, 2.0f, eg, rowstart, tid);
            __syncthreads();
            float* t = cur; cur = oth; oth = t;
        }
        wmma_term(cur, (const v16h*)(packB + k * 256), accE, accO, swave, lane);
    }
    __syncthreads();
    // merge even/odd chains, bias + relu, store to bufA
    const int m0 = lane & 15, half = lane >> 4;
    const int ntw = swave & 3, mtb = swave >> 2;
    const int n = ntw * 16 + m0;
    const float bn = bias[n];
#pragma unroll
    for (int i = 0; i < 3; ++i) {
        int mt = mtb + 2 * i;
        v8f c = accE[i] + accO[i];
#pragma unroll
        for (int r = 0; r < 8; ++r) {
            int m = mt * 16 + r + 8 * half;
            float v = c[r] + bn;
            bufA[m * H + n] = v > 0.0f ? v : 0.0f;
        }
    }
    if (swave < 4) {
        v8f c = accE[3] + accO[3];
#pragma unroll
        for (int r = 0; r < 8; ++r) {
            int m = 6 * 16 + r + 8 * half;
            float v = c[r] + bn;
            bufA[m * H + n] = v > 0.0f ? v : 0.0f;
        }
    }
    __syncthreads();
    for (int i = tid; i < (PADR - NPG) * H; i += BLOCK) bufA[NPG * H + i] = 0.0f;
    __syncthreads();
}

__global__ __launch_bounds__(BLOCK)
void chebnet_fused(const float* __restrict__ x,
                   const int* __restrict__ esrc, const int* __restrict__ edst,
                   const float* __restrict__ lambda_max,
                   const float* __restrict__ W1, const float* __restrict__ b1,
                   const float* __restrict__ b2, const float* __restrict__ b3,
                   const float* __restrict__ bn_gamma, const float* __restrict__ bn_beta,
                   const float* __restrict__ bn_mean, const float* __restrict__ bn_var,
                   const float* __restrict__ fc1_w, const float* __restrict__ fc1_b,
                   const float* __restrict__ fc2_w, const float* __restrict__ fc2_b,
                   const _Float16* __restrict__ packW, int2* __restrict__ edge_ws,
                   float* __restrict__ out, int epg) {
    __shared__ float bufA[PADR * H];
    __shared__ float bufB[PADR * H];
    __shared__ float dis[PADR];
    __shared__ int   rowstart[NPG + 4];
    __shared__ int   rcnt[PADR];

    const int g = blockIdx.x;
    const int tid = threadIdx.x;
    const int gbase = g * NPG;
    const int* es = esrc + (long)g * epg;
    const int* ed = edst + (long)g * epg;
    int2* eg = edge_ws + (long)g * epg;

    const float lmax = lambda_max[g];
    const float s  = -2.0f / lmax;
    const float cl =  2.0f / lmax - 1.0f;

    // ---- degree, dis = deg^{-1/2}, CSR build with precomputed coefficients ----
    if (tid < PADR) rcnt[tid] = 0;
    __syncthreads();
    for (int e = tid; e < epg; e += BLOCK) atomicAdd(&rcnt[es[e] - gbase], 1);
    __syncthreads();
    if (tid < PADR) {
        int d = rcnt[tid];
        dis[tid] = (d > 0) ? rsqrtf((float)d) : 0.0f;
    }
    if (tid == 0) {
        int a = 0;
        for (int i = 0; i < NPG; ++i) { rowstart[i] = a; a += rcnt[i]; }
        rowstart[NPG] = a;
    }
    __syncthreads();
    if (tid < NPG) rcnt[tid] = rowstart[tid];    // cursors
    __syncthreads();
    for (int e = tid; e < epg; e += BLOCK) {
        int r = es[e] - gbase;
        int c = ed[e] - gbase;
        int slot = atomicAdd(&rcnt[r], 1);
        float coef = s * dis[r] * dis[c];
        eg[slot] = make_int2(c, __float_as_int(coef));
    }
    __syncthreads();

    // ---- layer 1 (3 -> 64), scalar (K too small for WMMA) ----
    {
        float* t0 = bufB;          // 112 x 4
        float* t1 = bufB + 512;    // 112 x 4
        for (int i = tid; i < PADR * 4; i += BLOCK) { t0[i] = 0.0f; t1[i] = 0.0f; }
        for (int i = tid; i < PADR * H; i += BLOCK) bufA[i] = 0.0f;
        __syncthreads();
        for (int i = tid; i < NPG * 3; i += BLOCK) {
            int n = i / 3, c = i % 3;
            t0[n * 4 + c] = x[(long)(gbase + n) * 3 + c];
        }
        __syncthreads();
        float* cur = t0; float* oth = t1;
        for (int k = 0; k < 5; ++k) {
            if (k == 1) {
                gather3(oth, cur, cl, 1.0f, eg, rowstart, tid);   // oth pre-zeroed
                __syncthreads();
                float* t = cur; cur = oth; oth = t;
            } else if (k >= 2) {
                for (int i = tid; i < PADR * 4; i += BLOCK) oth[i] *= -0.5f;
                __syncthreads();
                gather3(oth, cur, cl, 2.0f, eg, rowstart, tid);
                __syncthreads();
                float* t = cur; cur = oth; oth = t;
            }
            const float* Wk = W1 + k * 3 * H;
            for (int i = tid; i < PADR * H; i += BLOCK) {
                int n = i >> 6, h = i & 63;
                bufA[i] += cur[n * 4 + 0] * Wk[h]
                         + cur[n * 4 + 1] * Wk[64 + h]
                         + cur[n * 4 + 2] * Wk[128 + h];
            }
            __syncthreads();
        }
        for (int i = tid; i < NPG * H; i += BLOCK) {
            float v = bufA[i] + b1[i & 63];
            bufA[i] = v > 0.0f ? v : 0.0f;
        }
        __syncthreads();
    }

    // ---- layers 2 & 3 (64 -> 64), WMMA ----
    cheb_layer64(bufA, bufB, eg, rowstart, cl,
                 (const v16h*)(packW),              b2);
    cheb_layer64(bufA, bufB, eg, rowstart, cl,
                 (const v16h*)(packW + PKB_HALVES), b3);

    // ---- global mean pool + BN + FC head ----
    {
        int grp = tid >> 6, f = tid & 63;
        float sum = 0.0f;
        for (int i = grp; i < NPG; i += 4) sum += bufA[i * H + f];
        bufB[tid] = sum;
        __syncthreads();
        if (tid < H) {
            float gm = (bufB[tid] + bufB[64 + tid] + bufB[128 + tid] + bufB[192 + tid])
                       * (1.0f / NPG);
            float v = (gm - bn_mean[tid]) * rsqrtf(bn_var[tid] + 1e-5f)
                      * bn_gamma[tid] + bn_beta[tid];
            bufB[256 + tid] = v;
        }
        __syncthreads();
        if (tid < 32) {
            float z = fc1_b[tid];
            for (int f2 = 0; f2 < H; ++f2) z += bufB[256 + f2] * fc1_w[f2 * 32 + tid];
            bufB[320 + tid] = z > 0.0f ? z : 0.0f;
        }
        __syncthreads();
        if (tid < 10) {
            float o = fc2_b[tid];
            for (int j = 0; j < 32; ++j) o += bufB[320 + j] * fc2_w[j * 10 + tid];
            bufB[352 + tid] = o;
        }
        __syncthreads();
        if (tid == 0) {
            float mx = bufB[352];
            for (int c = 1; c < 10; ++c) mx = fmaxf(mx, bufB[352 + c]);
            float se = 0.0f;
            for (int c = 0; c < 10; ++c) se += __expf(bufB[352 + c] - mx);
            bufB[363] = mx + __logf(se);
        }
        __syncthreads();
        if (tid < 10) out[(long)g * 10 + tid] = bufB[352 + tid] - bufB[363];
    }
}

extern "C" void kernel_launch(void* const* d_in, const int* in_sizes, int n_in,
                              void* d_out, int out_size, void* d_ws, size_t ws_size,
                              hipStream_t stream) {
    (void)n_in; (void)ws_size; (void)out_size;
    const float* x        = (const float*)d_in[0];
    const int*   ei       = (const int*)  d_in[1];
    const float* lmax     = (const float*)d_in[3];
    const float* W1       = (const float*)d_in[4];
    const float* b1       = (const float*)d_in[5];
    const float* W2       = (const float*)d_in[6];
    const float* b2       = (const float*)d_in[7];
    const float* W3       = (const float*)d_in[8];
    const float* b3       = (const float*)d_in[9];
    const float* bn_gamma = (const float*)d_in[10];
    const float* bn_beta  = (const float*)d_in[11];
    const float* bn_mean  = (const float*)d_in[12];
    const float* bn_var   = (const float*)d_in[13];
    const float* fc1_w    = (const float*)d_in[14];
    const float* fc1_b    = (const float*)d_in[15];
    const float* fc2_w    = (const float*)d_in[16];
    const float* fc2_b    = (const float*)d_in[17];

    const int E   = in_sizes[1] / 2;
    const int G   = in_sizes[3];
    const int epg = E / G;
    const int* esrc = ei;
    const int* edst = ei + E;

    _Float16* packW  = (_Float16*)d_ws;
    int2*     edgews = (int2*)((char*)d_ws + EDGE_OFS);

    pack_weights<<<(2 * PKB_HALVES + BLOCK - 1) / BLOCK, BLOCK, 0, stream>>>(
        W2, W3, packW);

    chebnet_fused<<<G, BLOCK, 0, stream>>>(
        x, esrc, edst, lmax, W1, b1, b2, b3,
        bn_gamma, bn_beta, bn_mean, bn_var,
        fc1_w, fc1_b, fc2_w, fc2_b,
        packW, edgews, (float*)d_out, epg);
}
